// FusedMoEModule_84713934946394
// MI455X (gfx1250) — compile-verified, matrix operand order checked by reading, not run on hardware
//
#include <hip/hip_runtime.h>
#include <hip/hip_bf16.h>

// Problem constants (match reference)
#define T_TOK 4096
#define K_DIM 1024
#define H_DIM 256
#define N_UP  512          // 2*H
#define E_NUM 16
#define TOPK  2
#define CAP   8192         // worst-case routed pairs per expert = T*TOPK
#define M_TILE 32          // routed rows per workgroup (2 WMMA M-blocks)
#define MAX_TILES (T_TOK * TOPK / M_TILE)   // 256

typedef __attribute__((ext_vector_type(8)))  _Float16 v8h;
typedef __attribute__((ext_vector_type(16))) _Float16 v16h;
typedef __attribute__((ext_vector_type(8)))  float    v8f;

// ---------------------------------------------------------------------------
// 16-bit WMMA operand striping (ISA 7.12.2, 16x16x32): lane L (0..15) holds
// halves[0:8] = K k0..k0+7 and halves[8:16] = K k0+16..k0+23; lane L+16 holds
// the +8-shifted K groups. Caller passes q = row_ptr + k0 + laneHi*8.
// ---------------------------------------------------------------------------

// Operand already in f16 (LDS stage buffer or pre-converted f16 weights).
static __device__ __forceinline__ v16h load_h16(const _Float16* q) {
    v8h lo = *(const v8h*)(q + 0);
    v8h hi = *(const v8h*)(q + 16);
    return __builtin_shufflevector(lo, hi, 0,1,2,3,4,5,6,7,8,9,10,11,12,13,14,15);
}

// Operand in f32 global memory (fallback path): lanes L and L+16 together
// consume one full 128B line per K-step.
static __device__ __forceinline__ v16h load_b_f32(const float* __restrict__ q) {
    float4 f0 = *(const float4*)(q + 0);
    float4 f1 = *(const float4*)(q + 4);
    float4 f2 = *(const float4*)(q + 16);
    float4 f3 = *(const float4*)(q + 20);
    v16h r;
    r[0]  = (_Float16)f0.x; r[1]  = (_Float16)f0.y; r[2]  = (_Float16)f0.z; r[3]  = (_Float16)f0.w;
    r[4]  = (_Float16)f1.x; r[5]  = (_Float16)f1.y; r[6]  = (_Float16)f1.z; r[7]  = (_Float16)f1.w;
    r[8]  = (_Float16)f2.x; r[9]  = (_Float16)f2.y; r[10] = (_Float16)f2.z; r[11] = (_Float16)f2.w;
    r[12] = (_Float16)f3.x; r[13] = (_Float16)f3.y; r[14] = (_Float16)f3.z; r[15] = (_Float16)f3.w;
    return r;
}

// ---------------------------------------------------------------------------
// Kernel 0: zero output accumulator and per-expert counters
// ---------------------------------------------------------------------------
__global__ void moe_zero_kernel(float4* __restrict__ out4, int* __restrict__ counts) {
    size_t idx = (size_t)blockIdx.x * blockDim.x + threadIdx.x;   // exactly T*K/4 threads
    float4 z; z.x = 0.f; z.y = 0.f; z.z = 0.f; z.w = 0.f;
    out4[idx] = z;
    if (blockIdx.x == 0 && threadIdx.x < E_NUM) counts[threadIdx.x] = 0;
}

// ---------------------------------------------------------------------------
// Kernel 1: routing scatter — bucket (token, weight) pairs per expert
// ---------------------------------------------------------------------------
__global__ void moe_route_kernel(const long long* __restrict__ ids,
                                 const float* __restrict__ tw,
                                 int* __restrict__ counts,
                                 int* __restrict__ tok,
                                 float* __restrict__ wgt) {
    int i = blockIdx.x * blockDim.x + threadIdx.x;
    if (i >= T_TOK * TOPK) return;
    int e = (int)ids[i];
    int pos = atomicAdd(&counts[e], 1);
    tok[e * CAP + pos] = i >> 1;   // token index (TOPK == 2)
    wgt[e * CAP + pos] = tw[i];
}

// ---------------------------------------------------------------------------
// Kernel 1.5: one-shot f32 -> f16 weight down-conversion (loop-invariant work;
// halves all subsequent weight bytes moved and removes cvt from GEMM loops)
// ---------------------------------------------------------------------------
__global__ void moe_cvt_kernel(const float4* __restrict__ src,
                               _Float16* __restrict__ dst, int n4) {
    int i = blockIdx.x * blockDim.x + threadIdx.x;
    if (i >= n4) return;
    float4 v = src[i];
    _Float16* d = dst + (size_t)i * 4;
    d[0] = (_Float16)v.x; d[1] = (_Float16)v.y; d[2] = (_Float16)v.z; d[3] = (_Float16)v.w;
}

// ---------------------------------------------------------------------------
// Kernel 2: routed up-GEMM + SwiGLU + down-GEMM + weighted combine
// grid = (MAX_TILES, E), block = 256 (8 wave32), 80KB dynamic LDS
// F16W: weights pre-converted to f16 in workspace
// ---------------------------------------------------------------------------
template <bool F16W>
__global__ __launch_bounds__(256) void moe_gemm_kernel(
    const float* __restrict__ X,    // [T, K] f32
    const void*  __restrict__ W1v,  // [E, 2H, K] f32 or f16
    const void*  __restrict__ W2v,  // [E, K, H]  f32 or f16
    const int*   __restrict__ counts,
    const int*   __restrict__ tok,
    const float* __restrict__ wgt,
    float*       __restrict__ out)  // [T, K] f32
{
    const int e    = blockIdx.y;
    const int tile = blockIdx.x;
    const int cnt  = counts[e];
    if (tile * M_TILE >= cnt) return;

    const int tid    = threadIdx.x;
    const int wid    = tid >> 5;
    const int lane   = tid & 31;
    const int laneLo = lane & 15;
    const int laneHi = lane >> 4;

    // dynamic LDS: [0,64KB) f16 A-stage, aliased later as f32 up-results;
    //              [64KB,80KB) f16 activation buffer
    extern __shared__ __align__(16) unsigned char smem[];
    _Float16 (*sA)[K_DIM]   = (_Float16(*)[K_DIM])smem;
    float    (*sUp)[N_UP]   = (float(*)[N_UP])smem;
    _Float16 (*sAct)[H_DIM] = (_Float16(*)[H_DIM])(smem + (size_t)M_TILE * K_DIM * 2);
    __shared__ int   sTok[M_TILE];
    __shared__ float sW[M_TILE];

    if (tid < M_TILE) {
        int idx = tile * M_TILE + tid;
        if (idx < cnt) { sTok[tid] = tok[e * CAP + idx]; sW[tid] = wgt[e * CAP + idx]; }
        else           { sTok[tid] = -1;                 sW[tid] = 0.f; }
    }
    __syncthreads();

    // ---- stage gathered hidden_states rows as f16 into LDS ----
    for (int i = tid; i < M_TILE * (K_DIM / 4); i += 256) {
        int m  = i >> 8;              // K_DIM/4 == 256 float4 per row
        int c4 = i & 255;
        int t  = sTok[m]; if (t < 0) t = 0;   // padded rows read row 0, masked later
        float4 v = *(const float4*)(X + (size_t)t * K_DIM + (size_t)c4 * 4);
        _Float16* d = &sA[m][c4 * 4];
        d[0] = (_Float16)v.x; d[1] = (_Float16)v.y; d[2] = (_Float16)v.z; d[3] = (_Float16)v.w;
    }
    __syncthreads();

    const float*    W1f = (const float*)W1v;
    const _Float16* W1h = (const _Float16*)W1v;
    const float*    W2f = (const float*)W2v;
    const _Float16* W2h = (const _Float16*)W2v;

    // ---- up-GEMM: [32 x 1024] * [1024 x 512] ; wave owns 64 cols x 2 M-blocks ----
    const int nb = wid * 64;
    v8f zero8 = {};
    v8f acc[2][4];
    #pragma unroll
    for (int mi = 0; mi < 2; ++mi)
        #pragma unroll
        for (int j = 0; j < 4; ++j) acc[mi][j] = zero8;

    for (int kt = 0; kt < K_DIM / 32; ++kt) {
        const int k0 = kt * 32;
        v16h a0 = load_h16(&sA[laneLo][k0 + laneHi * 8]);
        v16h a1 = load_h16(&sA[16 + laneLo][k0 + laneHi * 8]);
        #pragma unroll
        for (int j = 0; j < 4; ++j) {
            const int n = nb + j * 16 + laneLo;
            v16h b;
            if constexpr (F16W) b = load_h16(W1h + (size_t)e * N_UP * K_DIM + (size_t)n * K_DIM + k0 + laneHi * 8);
            else                b = load_b_f32(W1f + (size_t)e * N_UP * K_DIM + (size_t)n * K_DIM + k0 + laneHi * 8);
            acc[0][j] = __builtin_amdgcn_wmma_f32_16x16x32_f16(
                false, a0, false, b, (short)0, acc[0][j], false, false);
            acc[1][j] = __builtin_amdgcn_wmma_f32_16x16x32_f16(
                false, a1, false, b, (short)0, acc[1][j], false, false);
        }
    }
    __syncthreads();   // everyone done reading sA; safe to alias as sUp

    // spill up results (D layout: VGPR r -> row r + laneHi*8, col = laneLo)
    #pragma unroll
    for (int mi = 0; mi < 2; ++mi)
        #pragma unroll
        for (int j = 0; j < 4; ++j)
            #pragma unroll
            for (int r = 0; r < 8; ++r)
                sUp[mi * 16 + r + laneHi * 8][nb + j * 16 + laneLo] = acc[mi][j][r];
    __syncthreads();

    // ---- SwiGLU: act[m,h] = silu(up[m,h]) * up[m,h+H] ----
    for (int i = tid; i < M_TILE * H_DIM; i += 256) {
        int m = i >> 8;               // H_DIM == 256
        int h = i & 255;
        float g = sUp[m][h];
        float u = sUp[m][h + H_DIM];
        float s = g / (1.f + __expf(-g));
        sAct[m][h] = (_Float16)(s * u);
    }
    __syncthreads();

    // ---- down-GEMM: [32 x 256] * [256 x 1024] ; wave owns 128 cols x 2 M-blocks ----
    const int nb2 = wid * 128;
    v8f acc2[2][8];
    #pragma unroll
    for (int mi = 0; mi < 2; ++mi)
        #pragma unroll
        for (int j = 0; j < 8; ++j) acc2[mi][j] = zero8;

    #pragma unroll
    for (int kt = 0; kt < H_DIM / 32; ++kt) {
        const int k0 = kt * 32;
        v16h a0 = load_h16(&sAct[laneLo][k0 + laneHi * 8]);
        v16h a1 = load_h16(&sAct[16 + laneLo][k0 + laneHi * 8]);
        #pragma unroll
        for (int j = 0; j < 8; ++j) {
            const int n = nb2 + j * 16 + laneLo;
            v16h b;
            if constexpr (F16W) b = load_h16(W2h + (size_t)e * K_DIM * H_DIM + (size_t)n * H_DIM + k0 + laneHi * 8);
            else                b = load_b_f32(W2f + (size_t)e * K_DIM * H_DIM + (size_t)n * H_DIM + k0 + laneHi * 8);
            acc2[0][j] = __builtin_amdgcn_wmma_f32_16x16x32_f16(
                false, a0, false, b, (short)0, acc2[0][j], false, false);
            acc2[1][j] = __builtin_amdgcn_wmma_f32_16x16x32_f16(
                false, a1, false, b, (short)0, acc2[1][j], false, false);
        }
    }

    // ---- epilogue: weight-scale + atomic combine into out[t, :] ----
    #pragma unroll
    for (int mi = 0; mi < 2; ++mi) {
        #pragma unroll
        for (int r = 0; r < 8; ++r) {
            const int m = mi * 16 + r + laneHi * 8;
            const int t = sTok[m];
            if (t < 0) continue;
            const float w = sW[m];
            #pragma unroll
            for (int j = 0; j < 8; ++j) {
                const int n = nb2 + j * 16 + laneLo;
                atomicAdd(out + (size_t)t * K_DIM + n, w * acc2[mi][j][r]);
            }
        }
    }
}

// ---------------------------------------------------------------------------
// Host launcher
// ---------------------------------------------------------------------------
extern "C" void kernel_launch(void* const* d_in, const int* in_sizes, int n_in,
                              void* d_out, int out_size, void* d_ws, size_t ws_size,
                              hipStream_t stream) {
    const float*     X  = (const float*)d_in[0];      // hidden_states [T,K] f32
    const float*     tw = (const float*)d_in[1];      // topk_weights  [T,2] f32
    const long long* ti = (const long long*)d_in[2];  // topk_ids      [T,2] i64
    const float*     W1 = (const float*)d_in[3];      // up_weight   [E,2H,K] f32
    const float*     W2 = (const float*)d_in[4];      // down_weight [E,K,H]  f32
    float*           out = (float*)d_out;             // [T,K] f32

    // workspace layout: counts | tok buckets | wgt buckets | (optional) f16 weights
    int*   counts = (int*)d_ws;
    int*   tok    = (int*)((char*)d_ws + 256);
    float* wgt    = (float*)((char*)d_ws + 256 + (size_t)E_NUM * CAP * sizeof(int));

    const size_t routeBytes = 256 + (size_t)E_NUM * CAP * (sizeof(int) + sizeof(float));
    const size_t w1Elems = (size_t)E_NUM * N_UP * K_DIM;    // 8M
    const size_t w2Elems = (size_t)E_NUM * K_DIM * H_DIM;   // 4M
    const size_t off1 = (routeBytes + 4095) & ~(size_t)4095;
    const size_t off2 = off1 + w1Elems * sizeof(_Float16);
    const size_t need = off2 + w2Elems * sizeof(_Float16);
    const bool   f16w = (ws_size >= need);
    _Float16* W1h = (_Float16*)((char*)d_ws + off1);
    _Float16* W2h = (_Float16*)((char*)d_ws + off2);

    const size_t smemBytes = (size_t)M_TILE * K_DIM * 2 + (size_t)M_TILE * H_DIM * 2; // 80KB

    // 1) zero output + counters
    moe_zero_kernel<<<(T_TOK * K_DIM / 4) / 256, 256, 0, stream>>>((float4*)out, counts);

    // 2) routing scatter
    moe_route_kernel<<<(T_TOK * TOPK + 255) / 256, 256, 0, stream>>>(ti, tw, counts, tok, wgt);

    // 3) fused routed MoE (up GEMM -> SwiGLU -> down GEMM -> combine)
    if (f16w) {
        moe_cvt_kernel<<<(int)((w1Elems / 4 + 255) / 256), 256, 0, stream>>>((const float4*)W1, W1h, (int)(w1Elems / 4));
        moe_cvt_kernel<<<(int)((w2Elems / 4 + 255) / 256), 256, 0, stream>>>((const float4*)W2, W2h, (int)(w2Elems / 4));
        moe_gemm_kernel<true><<<dim3(MAX_TILES, E_NUM), 256, smemBytes, stream>>>(
            X, W1h, W2h, counts, tok, wgt, out);
    } else {
        moe_gemm_kernel<false><<<dim3(MAX_TILES, E_NUM), 256, smemBytes, stream>>>(
            X, W1, W2, counts, tok, wgt, out);
    }

    (void)in_sizes; (void)n_in; (void)out_size;
}
